// GNN_70935679860797
// MI455X (gfx1250) — matrix operand order repaired and checked by reading, not compiled, not run on hardware
//
#include <hip/hip_runtime.h>
#include <math.h>

// ---------------------------------------------------------------------------
// Types for CDNA5 WMMA (wave32)
// ---------------------------------------------------------------------------
typedef __attribute__((ext_vector_type(16))) __bf16 v16bf;
typedef __attribute__((ext_vector_type(8)))  float  v8f;

#define N_NODES 100000
#define D_IN    512
#define D_HID   128
#define D_OUT   40
#define D_OUTP  48   // padded to 3 WMMA n-tiles

static __device__ __forceinline__ __bf16 f2bf(float f) {
    unsigned u = __builtin_bit_cast(unsigned, f);
    unsigned r = (u + 0x7FFFu + ((u >> 16) & 1u)) >> 16;  // RNE truncation
    unsigned short s = (unsigned short)r;
    return __builtin_bit_cast(__bf16, s);
}

static __device__ __forceinline__ unsigned pack2bf(float a, float b) {
    unsigned lo = __builtin_bit_cast(unsigned short, f2bf(a));
    unsigned hi = __builtin_bit_cast(unsigned short, f2bf(b));
    return lo | (hi << 16);
}

// A-fragment slot->K mapping (16-bit A 16x32, wave32):
//   lane L: row = L&15, hi = L>>4
//   slot pair j=2v:   v<4 -> kk = 2v + 8*hi ; v>=4 -> kk = 16 + 2(v-4) + 8*hi
static __device__ __forceinline__ int a_kk_of_v(int v, int hi) {
    return (v < 4) ? (2 * v + 8 * hi) : (16 + 2 * (v - 4) + 8 * hi);
}

// ---------------------------------------------------------------------------
// gcn_norm prep
// ---------------------------------------------------------------------------
__global__ void k_init_deg(float* deg, int n) {
    int i = blockIdx.x * blockDim.x + threadIdx.x;
    if (i < n) deg[i] = 1.0f;                 // self-loop contribution
}

__global__ void k_deg_accum(const int* __restrict__ dst, float* deg, int E) {
    int e = blockIdx.x * blockDim.x + threadIdx.x;
    if (e < E) atomicAdd(&deg[dst[e]], 1.0f);
}

__global__ void k_dinv(const float* __restrict__ deg, float* dinv, int n) {
    int i = blockIdx.x * blockDim.x + threadIdx.x;
    if (i < n) {
        float d = deg[i];
        dinv[i] = (d > 0.0f) ? rsqrtf(d) : 0.0f;
    }
}

__global__ void k_norm(const int* __restrict__ src, const int* __restrict__ dst,
                       const float* __restrict__ dinv, float* norm, int E) {
    int e = blockIdx.x * blockDim.x + threadIdx.x;
    if (e < E) norm[e] = dinv[src[e]] * dinv[dst[e]];
}

// ---------------------------------------------------------------------------
// Weight conversion into B-fragment order: [n_tile][k_step][lane][16] bf16
//   B layout (16-bit B 32x16): lane L: col = L&15, hi = L>>4 ; slot j -> kk = j<8 ? j + 8*hi : 16 + (j-8) + 8*hi
//   (pairs j=2v,2v+1 <-> kk,kk+1 contiguous in K)
// ---------------------------------------------------------------------------
__global__ void k_cvt_w1_frag(const float* __restrict__ W1, unsigned* W1f) {
    int t = blockIdx.x * blockDim.x + threadIdx.x;       // dword index
    if (t >= (D_IN / 32) * (D_HID / 16) * 32 * 8) return; // 32768
    int v    = t & 7;
    int lane = (t >> 3) & 31;
    int ks   = (t >> 8) & 15;
    int nt   = t >> 12;
    int col = lane & 15, hi = lane >> 4;
    int k = ks * 32 + a_kk_of_v(v, hi);
    int n = nt * 16 + col;
    W1f[t] = pack2bf(W1[k * D_HID + n], W1[(k + 1) * D_HID + n]);
}

__global__ void k_cvt_w2_frag(const float* __restrict__ W2, unsigned* W2f) {
    int t = blockIdx.x * blockDim.x + threadIdx.x;       // dword index
    if (t >= 3 * 4 * 32 * 8) return;                     // 3072
    int v    = t & 7;
    int lane = (t >> 3) & 31;
    int ks   = (t >> 8) & 3;
    int nt   = t >> 10;
    int col = lane & 15, hi = lane >> 4;
    int k = ks * 32 + a_kk_of_v(v, hi);
    int n = nt * 16 + col;
    float a = (n < D_OUT) ? W2[k * D_OUT + n] : 0.0f;
    float b = (n < D_OUT) ? W2[(k + 1) * D_OUT + n] : 0.0f;
    W2f[t] = pack2bf(a, b);
}

// ---------------------------------------------------------------------------
// GEMM1: h1[N,128] = x[N,512] @ W1[512,128]   (bf16 WMMA, f32 accum)
// A strip (16x512) converted+swizzled into LDS in fragment order; 8 waves
// each own one 16-col tile; inner loop = 2x ds_load_b128 + 2x global_load_b128
// + v_wmma per k-step.
// ---------------------------------------------------------------------------
__global__ __launch_bounds__(256) void k_gemm1(const float* __restrict__ x,
                                               const __bf16* __restrict__ W1f,
                                               float* __restrict__ h1) {
    __shared__ v16bf As[16 * 32];                 // [k_step][lane], 16 KB
    const int m0 = blockIdx.x * 16;

    // stage + swizzle: thread handles one K-pair (one dword of bf16)
    unsigned* Asu = (unsigned*)As;
    for (int i = threadIdx.x; i < 16 * (D_IN / 2); i += 256) {
        int r  = i >> 8;                 // row in strip
        int k  = (i & 255) * 2;          // even K
        int ks = k >> 5;
        int kk = k & 31;
        int hi = ((kk & 15) >= 8) ? 1 : 0;
        int lane = r + 16 * hi;
        int j = ((kk >= 16) ? 8 : 0) + ((kk & 15) - 8 * hi);  // even slot
        const float* px = x + (size_t)(m0 + r) * D_IN + k;
        Asu[(ks * 32 + lane) * 8 + (j >> 1)] = pack2bf(px[0], px[1]);
    }
    __syncthreads();

    const int wave = threadIdx.x >> 5;
    const int lane = threadIdx.x & 31;
    const v16bf* B = (const v16bf*)W1f + (size_t)wave * 16 * 32;  // [ks][lane]

    v8f acc = {};
#pragma unroll 4
    for (int ks = 0; ks < 16; ++ks) {
        v16bf a = As[ks * 32 + lane];
        v16bf b = B[ks * 32 + lane];
        acc = __builtin_amdgcn_wmma_f32_16x16x32_bf16(false, a, false, b,
                                                      (short)0, acc, false, false);
    }

    const int col = lane & 15, hi = lane >> 4, n0 = wave * 16;
#pragma unroll
    for (int v = 0; v < 8; ++v)
        h1[(size_t)(m0 + v + 8 * hi) * D_HID + n0 + col] = acc[v];
}

// ---------------------------------------------------------------------------
// Layer-1 aggregation: out1 = b1 + dinv^2 * h1 (self loop), then edge atomics
// ---------------------------------------------------------------------------
__global__ void k_agg1_init(const float* __restrict__ b1,
                            const float* __restrict__ dinv,
                            const float* __restrict__ h1,
                            float* out1) {
    long long i = (long long)blockIdx.x * blockDim.x + threadIdx.x;
    if (i < (long long)N_NODES * D_HID) {
        int node = (int)(i >> 7);
        int f    = (int)i & (D_HID - 1);
        float dv = dinv[node];
        out1[i] = b1[f] + dv * dv * h1[i];
    }
}

__global__ void k_agg1_edges(const int* __restrict__ src, const int* __restrict__ dst,
                             const float* __restrict__ norm,
                             const float* __restrict__ h1,
                             float* out1, int E) {
    long long idx = (long long)blockIdx.x * blockDim.x + threadIdx.x;
    if (idx >= (long long)E * 32) return;
    int e = (int)(idx >> 5);
    int f = ((int)idx & 31) * 4;
    if (f == 0 && e + 4096 < E) {
        __builtin_prefetch(&src[e + 4096], 0, 1);   // global_prefetch_b8
        __builtin_prefetch(&norm[e + 4096], 0, 1);
    }
    int s = src[e], d = dst[e];
    float nv = norm[e];
    const float4 hv = *(const float4*)(h1 + (long long)s * D_HID + f);
    float* o = out1 + (long long)d * D_HID + f;
    atomicAdd(o + 0, nv * hv.x);
    atomicAdd(o + 1, nv * hv.y);
    atomicAdd(o + 2, nv * hv.z);
    atomicAdd(o + 3, nv * hv.w);
}

// ReLU + convert + swizzle into A-fragment order for GEMM2:
//   hf layout [m_tile][k_step(4)][lane(32)][16] bf16, dword-granular writes
__global__ void k_relu_cvt_frag(const float* __restrict__ out1, unsigned* hf) {
    long long t = (long long)blockIdx.x * blockDim.x + threadIdx.x;
    if (t >= (long long)N_NODES * D_HID / 2) return;
    int v    = (int)t & 7;
    int lane = ((int)(t >> 3)) & 31;
    int ks   = ((int)(t >> 8)) & 3;
    long long mt = t >> 10;
    int row = (int)mt * 16 + (lane & 15);
    int hi  = lane >> 4;
    int k   = ks * 32 + a_kk_of_v(v, hi);
    const float* p = out1 + (size_t)row * D_HID + k;
    hf[t] = pack2bf(fmaxf(p[0], 0.0f), fmaxf(p[1], 0.0f));
}

// ---------------------------------------------------------------------------
// GEMM2: h2[N,40] = relu_h[N,128] @ W2[128,40]  (padded N=48, 3 waves/strip)
// Both operands pre-swizzled: inner loop = 4x global_load_b128 + v_wmma.
// ---------------------------------------------------------------------------
__global__ __launch_bounds__(96) void k_gemm2(const __bf16* __restrict__ hf,
                                              const __bf16* __restrict__ W2f,
                                              float* __restrict__ h2) {
    const int mt   = blockIdx.x;
    const int wave = threadIdx.x >> 5;    // 0..2
    const int lane = threadIdx.x & 31;

    const v16bf* A = (const v16bf*)hf  + (size_t)mt * 4 * 32;    // [ks][lane]
    const v16bf* B = (const v16bf*)W2f + (size_t)wave * 4 * 32;  // [ks][lane]

    v8f acc = {};
#pragma unroll
    for (int ks = 0; ks < 4; ++ks) {
        v16bf a = A[ks * 32 + lane];
        v16bf b = B[ks * 32 + lane];
        acc = __builtin_amdgcn_wmma_f32_16x16x32_bf16(false, a, false, b,
                                                      (short)0, acc, false, false);
    }

    const int col = lane & 15, hi = lane >> 4;
    const int m0 = mt * 16, c = wave * 16 + col;
    if (c < D_OUT) {
#pragma unroll
        for (int v = 0; v < 8; ++v)
            h2[(long long)(m0 + v + 8 * hi) * D_OUT + c] = acc[v];
    }
}

// ---------------------------------------------------------------------------
// Layer-2 aggregation into d_out, then in-place log_softmax
// ---------------------------------------------------------------------------
__global__ void k_agg2_init(const float* __restrict__ b2,
                            const float* __restrict__ dinv,
                            const float* __restrict__ h2,
                            float* out) {
    long long i = (long long)blockIdx.x * blockDim.x + threadIdx.x;
    if (i < (long long)N_NODES * D_OUT) {
        int node = (int)(i / D_OUT);
        int f    = (int)(i % D_OUT);
        float dv = dinv[node];
        out[i] = b2[f] + dv * dv * h2[i];
    }
}

__global__ void k_agg2_edges(const int* __restrict__ src, const int* __restrict__ dst,
                             const float* __restrict__ norm,
                             const float* __restrict__ h2,
                             float* out, int E) {
    long long idx = (long long)blockIdx.x * blockDim.x + threadIdx.x;
    if (idx >= (long long)E * 16) return;
    int e = (int)(idx >> 4);
    int g = (int)idx & 15;
    if (g >= 10) return;                       // 40 feats = 10 float4 groups
    int f = g * 4;
    int s = src[e], d = dst[e];
    float nv = norm[e];
    const float4 hv = *(const float4*)(h2 + (long long)s * D_OUT + f);
    float* o = out + (long long)d * D_OUT + f;
    atomicAdd(o + 0, nv * hv.x);
    atomicAdd(o + 1, nv * hv.y);
    atomicAdd(o + 2, nv * hv.z);
    atomicAdd(o + 3, nv * hv.w);
}

__global__ void k_log_softmax(float* out, int n) {
    int i = blockIdx.x * blockDim.x + threadIdx.x;
    if (i >= n) return;
    float* p = out + (long long)i * D_OUT;
    float m = -INFINITY;
#pragma unroll
    for (int j = 0; j < D_OUT; ++j) m = fmaxf(m, p[j]);
    float s = 0.0f;
#pragma unroll
    for (int j = 0; j < D_OUT; ++j) s += __expf(p[j] - m);
    float l = m + __logf(s);
#pragma unroll
    for (int j = 0; j < D_OUT; ++j) p[j] = p[j] - l;
}

// ---------------------------------------------------------------------------
// Launch
// ---------------------------------------------------------------------------
extern "C" void kernel_launch(void* const* d_in, const int* in_sizes, int n_in,
                              void* d_out, int out_size, void* d_ws, size_t ws_size,
                              hipStream_t stream) {
    const float* x  = (const float*)d_in[0];
    const int*   ei = (const int*)d_in[1];
    const float* W1 = (const float*)d_in[2];
    const float* b1 = (const float*)d_in[3];
    const float* W2 = (const float*)d_in[4];
    const float* b2 = (const float*)d_in[5];
    float* out = (float*)d_out;

    const int N = in_sizes[0] / D_IN;       // 100000
    const int E = in_sizes[1] / 2;          // 1600000
    const int* src = ei;
    const int* dst = ei + E;

    // workspace layout (256-B aligned slices)
    char* base = (char*)d_ws;
    size_t off = 0;
    auto carve = [&](size_t bytes) -> char* {
        char* p = base + off;
        off = (off + bytes + 255) & ~(size_t)255;
        return p;
    };
    float*  deg  = (float*)carve((size_t)N * 4);
    float*  dinv = (float*)carve((size_t)N * 4);
    float*  nrm  = (float*)carve((size_t)E * 4);
    __bf16* W1f  = (__bf16*)carve((size_t)D_IN * D_HID * 2);
    __bf16* W2f  = (__bf16*)carve((size_t)D_HID * D_OUTP * 2);
    float*  h1   = (float*)carve((size_t)N * D_HID * 4);
    float*  out1 = (float*)carve((size_t)N * D_HID * 4);
    __bf16* hf   = (__bf16*)carve((size_t)N * D_HID * 2);
    float*  h2   = (float*)carve((size_t)N * D_OUT * 4);

    const int T = 256;
    auto blk = [](long long n, int t) { return (unsigned)((n + t - 1) / t); };

    // norm prep
    k_init_deg  <<<blk(N, T), T, 0, stream>>>(deg, N);
    k_deg_accum <<<blk(E, T), T, 0, stream>>>(dst, deg, E);
    k_dinv      <<<blk(N, T), T, 0, stream>>>(deg, dinv, N);
    k_norm      <<<blk(E, T), T, 0, stream>>>(src, dst, dinv, nrm, E);

    // weights -> bf16 B-fragment order
    k_cvt_w1_frag <<<blk(32768, T), T, 0, stream>>>(W1, (unsigned*)W1f);
    k_cvt_w2_frag <<<blk(3072, T), T, 0, stream>>>(W2, (unsigned*)W2f);

    // layer 1
    k_gemm1 <<<N / 16, 256, 0, stream>>>(x, W1f, h1);
    k_agg1_init  <<<blk((long long)N * D_HID, T), T, 0, stream>>>(b1, dinv, h1, out1);
    k_agg1_edges <<<blk((long long)E * 32, T), T, 0, stream>>>(src, dst, nrm, h1, out1, E);
    k_relu_cvt_frag <<<blk((long long)N * D_HID / 2, T), T, 0, stream>>>(out1, (unsigned*)hf);

    // layer 2
    k_gemm2 <<<N / 16, 96, 0, stream>>>(hf, W2f, h2);
    k_agg2_init  <<<blk((long long)N * D_OUT, T), T, 0, stream>>>(b2, dinv, h2, out);
    k_agg2_edges <<<blk((long long)E * 16, T), T, 0, stream>>>(src, dst, nrm, h2, out, E);

    // output
    k_log_softmax <<<blk(N, T), T, 0, stream>>>(out, N);
}
